// backbone_pointnet2_76828374991458
// MI455X (gfx1250) — matrix operand order repaired and checked by reading, not compiled, not run on hardware
//
#include <hip/hip_runtime.h>
#include <hip/hip_bf16.h>
#include <math.h>

// ---------------------------------------------------------------------------
// PointNet++ backbone for gfx1250 (MI455X).
// 1x1-conv MLP layers run on the WMMA pipe (v_wmma_f32_16x16x32_f16, one
// wave32 per 16x64 tile, 4 accumulators). A and B operands are pre-swizzled
// into fragment-exact f16 layouts so the inner loop is 5 contiguous 32-byte
// loads + 4 WMMAs per K-block: no per-element converts, selects, or waits.
// BN is folded into the conv epilogue.
// ---------------------------------------------------------------------------

typedef __attribute__((ext_vector_type(16))) _Float16 v16h;
typedef __attribute__((ext_vector_type(8)))  float    v8f;

#define BN_EPS 1e-5f
#define NEG_SLOPE 0.2f

// ===========================================================================
// Operand swizzles (per CDNA5 ISA 7.12.2 fragment layouts, wave32)
// A fragment (16-bit, 16x32): lane (half_sel,l16) holds 16 halves; stored so
//   they are contiguous:  block[m][kbi][32] with k-groups in natural order {0,1,2,3}.
// B fragment (16-bit, 32x16): lane holds k = half_sel*16 + h at column n here (code's choice, not ISA table order);
//   stored as block[kbi][n][32] with k%32 in natural order.
// K tail is zero-padded so the conv kernel needs no bounds handling.
// ===========================================================================
__global__ __launch_bounds__(256)
void swizzle_w_kernel(const float* __restrict__ W, _Float16* __restrict__ Ws,
                      int Cout, int Cin, int KB)
{
    int tid = blockIdx.x * 256 + threadIdx.x;
    int total = Cout * KB * 32;
    if (tid >= total) return;
    int pos = tid & 31;
    int kbi = (tid >> 5) % KB;
    int m   = tid / (32 * KB);
    int grp  = pos >> 3;                              // position group 0..3
    int kgrp = grp;                                   // natural k order
    int k = kbi * 32 + kgrp * 8 + (pos & 7);
    float v = (k < Cin) ? W[(size_t)m * Cin + k] : 0.0f;
    Ws[tid] = (_Float16)v;
}

__global__ __launch_bounds__(256)
void swizzle_x_kernel(const float* __restrict__ X, _Float16* __restrict__ Xs,
                      int Cin, int P, int KB)
{
    size_t tid = (size_t)blockIdx.x * 256 + threadIdx.x;
    size_t total = (size_t)KB * 8 * P;
    if (tid >= total) return;
    int n   = (int)(tid % P);
    int kq  = (int)((tid / P) & 7);                   // quad of 4 k-values
    int kbi = (int)(tid / ((size_t)8 * P));
    int kbase = kbi * 32 + kq * 4;
    union { _Float16 h[4]; uint2 u; } pk;
#pragma unroll
    for (int i = 0; i < 4; ++i) {
        int k = kbase + i;
        float v = (k < Cin) ? X[(size_t)k * P + n] : 0.0f;
        pk.h[i] = (_Float16)v;
    }
    *(uint2*)(Xs + ((size_t)kbi * P + n) * 32 + kq * 4) = pk.u;
}

// ===========================================================================
// WMMA conv: Y = act(bn(W @ X + cbias)).  Operands pre-swizzled f16.
// Ws:[Cout][KB][32] halves, Xs:[KB][P][32] halves, Y:[Cout,P] f32.
// ACT: 0=none, 1=relu, 2=leaky(0.2).  Requires P % 64 == 0.
// ===========================================================================
template <int ACT, bool HAS_BIAS, bool HAS_BN>
__global__ __launch_bounds__(32)
void conv_bn_act_wmma(const _Float16* __restrict__ Ws,
                      const float* __restrict__ cbias,
                      const float* __restrict__ bg, const float* __restrict__ bb,
                      const float* __restrict__ bm, const float* __restrict__ bv,
                      const _Float16* __restrict__ Xs, float* __restrict__ Y,
                      int Cout, int KB, int P)
{
    const int lane = threadIdx.x;            // 0..31
    const int half_sel = lane >> 4;          // 0|1
    const int l16 = lane & 15;
    const int n0 = blockIdx.x * 64;          // P tile (4 x 16)
    const int m0 = blockIdx.y * 16;          // Cout tile

    const int am = m0 + l16;                              // A row (M)
    const int amc = (am < Cout) ? am : (Cout - 1);        // clamp: never stored
    const _Float16* __restrict__ wp = Ws + (size_t)amc * KB * 32 + half_sel * 16;
    const _Float16* __restrict__ xp = Xs + (size_t)(n0 + l16) * 32 + half_sel * 16;
    const size_t xstep = (size_t)P * 32;

    v8f acc0 = {}, acc1 = {}, acc2 = {}, acc3 = {};
    for (int kbi = 0; kbi < KB; ++kbi) {
        v16h a  = *(const v16h*)(wp);
        v16h b0 = *(const v16h*)(xp);
        v16h b1 = *(const v16h*)(xp + 512);    // +16 columns
        v16h b2 = *(const v16h*)(xp + 1024);
        v16h b3 = *(const v16h*)(xp + 1536);
        wp += 32;
        xp += xstep;
        acc0 = __builtin_amdgcn_wmma_f32_16x16x32_f16(false, a, false, b0, (short)0, acc0, false, false);
        acc1 = __builtin_amdgcn_wmma_f32_16x16x32_f16(false, a, false, b1, (short)0, acc1, false, false);
        acc2 = __builtin_amdgcn_wmma_f32_16x16x32_f16(false, a, false, b2, (short)0, acc2, false, false);
        acc3 = __builtin_amdgcn_wmma_f32_16x16x32_f16(false, a, false, b3, (short)0, acc3, false, false);
    }

    // epilogue: D f32 layout -> element (m = r + 8*half_sel, n = l16 + 16*t)
#pragma unroll
    for (int r = 0; r < 8; ++r) {
        int mm = m0 + r + half_sel * 8;
        if (mm < Cout) {
            float sc = 1.0f, bi = 0.0f;
            if (HAS_BN) {
                float rs = rsqrtf(bv[mm] + BN_EPS);
                sc = bg[mm] * rs;
                bi = bb[mm] - bg[mm] * bm[mm] * rs;
            }
            float cb = HAS_BIAS ? cbias[mm] : 0.0f;
            float* __restrict__ yrow = Y + (size_t)mm * P + n0 + l16;
            float y0 = acc0[r], y1 = acc1[r], y2 = acc2[r], y3 = acc3[r];
            if (HAS_BIAS) { y0 += cb; y1 += cb; y2 += cb; y3 += cb; }
            if (HAS_BN)   { y0 = y0 * sc + bi; y1 = y1 * sc + bi; y2 = y2 * sc + bi; y3 = y3 * sc + bi; }
            if (ACT == 1) {
                y0 = fmaxf(y0, 0.0f); y1 = fmaxf(y1, 0.0f);
                y2 = fmaxf(y2, 0.0f); y3 = fmaxf(y3, 0.0f);
            } else if (ACT == 2) {
                y0 = (y0 > 0.0f) ? y0 : NEG_SLOPE * y0;
                y1 = (y1 > 0.0f) ? y1 : NEG_SLOPE * y1;
                y2 = (y2 > 0.0f) ? y2 : NEG_SLOPE * y2;
                y3 = (y3 > 0.0f) ? y3 : NEG_SLOPE * y3;
            }
            yrow[0]  = y0;
            yrow[16] = y1;
            yrow[32] = y2;
            yrow[48] = y3;
        }
    }
}

// ===========================================================================
// Farthest point sampling: one workgroup per batch, iterative argmax.
// ===========================================================================
__global__ __launch_bounds__(256)
void fps_kernel(const float* __restrict__ xyz, int stride, int N, int npoint,
                int* __restrict__ out)
{
    __shared__ float sdist[4096];
    __shared__ float rval[256];
    __shared__ int   rarg[256];
    __shared__ int   sfar;
    __shared__ float cc[3];
    const int b = blockIdx.x;
    const int t = threadIdx.x;
    const float* pts = xyz + (size_t)b * N * stride;

    for (int j = t; j < N; j += 256) sdist[j] = 1e10f;
    if (t == 0) sfar = 0;
    __syncthreads();

    for (int i = 0; i < npoint; ++i) {
        if (t == 0) {
            int far = sfar;
            out[b * npoint + i] = far;
            cc[0] = pts[far * stride + 0];
            cc[1] = pts[far * stride + 1];
            cc[2] = pts[far * stride + 2];
        }
        __syncthreads();
        float cx = cc[0], cy = cc[1], cz = cc[2];
        float bestv = -1.0f; int besti = 0;
        for (int j = t; j < N; j += 256) {
            float dx = pts[j * stride + 0] - cx;
            float dy = pts[j * stride + 1] - cy;
            float dz = pts[j * stride + 2] - cz;
            float d = dx * dx + dy * dy + dz * dz;
            float nd = fminf(sdist[j], d);
            sdist[j] = nd;
            if (nd > bestv) { bestv = nd; besti = j; }
        }
        rval[t] = bestv; rarg[t] = besti;
        __syncthreads();
        for (int s = 128; s > 0; s >>= 1) {
            if (t < s) {
                if (rval[t + s] > rval[t] ||
                    (rval[t + s] == rval[t] && rarg[t + s] < rarg[t])) {
                    rval[t] = rval[t + s]; rarg[t] = rarg[t + s];
                }
            }
            __syncthreads();
        }
        if (t == 0) sfar = rarg[0];
        __syncthreads();
    }
}

// gather selected xyz -> new_xyz [B,S,3]
__global__ __launch_bounds__(256)
void gather_xyz_kernel(const float* __restrict__ xyz, int stride,
                       const int* __restrict__ idx, float* __restrict__ out,
                       int B, int N, int S)
{
    int tid = blockIdx.x * blockDim.x + threadIdx.x;
    if (tid >= B * S) return;
    int b = tid / S;
    int j = idx[tid];
    const float* p = xyz + ((size_t)b * N + j) * stride;
    float* o = out + (size_t)tid * 3;
    o[0] = p[0]; o[1] = p[1]; o[2] = p[2];
}

// ball query: first K indices (ascending) with d2 < r2, padded with first hit
__global__ __launch_bounds__(256)
void ball_query_kernel(const float* __restrict__ xyz, int stride,
                       const float* __restrict__ new_xyz, float r2, int K,
                       int* __restrict__ idx, int B, int N, int S)
{
    int tid = blockIdx.x * blockDim.x + threadIdx.x;
    if (tid >= B * S) return;
    int b = tid / S;
    const float* pts = xyz + (size_t)b * N * stride;
    const float* q = new_xyz + (size_t)tid * 3;
    float qx = q[0], qy = q[1], qz = q[2];
    int* o = idx + (size_t)tid * K;
    int cnt = 0, first = 0;
    for (int j = 0; j < N && cnt < K; ++j) {
        float dx = pts[j * stride + 0] - qx;
        float dy = pts[j * stride + 1] - qy;
        float dz = pts[j * stride + 2] - qz;
        if (dx * dx + dy * dy + dz * dz < r2) {
            if (cnt == 0) first = j;
            o[cnt++] = j;
        }
    }
    for (; cnt < K; ++cnt) o[cnt] = first;
}

// build grouped tensor G[3+Cf, B*S*K]: rows 0-2 centered xyz, rest gathered feats
__global__ __launch_bounds__(256)
void group_kernel(const float* __restrict__ xyz, int stride,
                  const float* __restrict__ new_xyz,
                  const float* __restrict__ feats, const int* __restrict__ idx,
                  float* __restrict__ G, int B, int N, int S, int K, int Cf)
{
    int tid = blockIdx.x * blockDim.x + threadIdx.x;
    size_t P = (size_t)B * S * K;
    if (tid >= (int)P) return;
    int s = (tid / K) % S;
    int b = tid / (S * K);
    int j = idx[tid];
    const float* p = xyz + ((size_t)b * N + j) * stride;
    const float* q = new_xyz + ((size_t)b * S + s) * 3;
    G[0 * P + tid] = p[0] - q[0];
    G[1 * P + tid] = p[1] - q[1];
    G[2 * P + tid] = p[2] - q[2];
    size_t BN = (size_t)B * N;
    for (int c = 0; c < Cf; ++c)
        G[(size_t)(3 + c) * P + tid] = feats[(size_t)c * BN + (size_t)b * N + j];
}

// max over K samples: X[C, BS*K] -> Y[C, BS]
__global__ __launch_bounds__(256)
void maxpool_kernel(const float* __restrict__ X, float* __restrict__ Y,
                    int C, int BS, int K)
{
    int tid = blockIdx.x * blockDim.x + threadIdx.x;
    if (tid >= C * BS) return;
    int c = tid / BS;
    int p = tid % BS;
    const float* x = X + ((size_t)c * BS + p) * K;
    float m = x[0];
    for (int k = 1; k < K; ++k) m = fmaxf(m, x[k]);
    Y[(size_t)c * BS + p] = m;
}

// broadcast global feature kf[C,B] -> out[C, B*n]
__global__ __launch_bounds__(256)
void broadcast_kernel(const float* __restrict__ kf, float* __restrict__ out,
                      int C, int B, int n)
{
    int tid = blockIdx.x * blockDim.x + threadIdx.x;
    int P = B * n;
    if (tid >= C * P) return;
    int p = tid % P;
    int c = tid / P;
    int b = p / n;
    out[tid] = kf[(size_t)c * B + b];
}

// 3-NN inverse-distance interpolation: kf[Ck, B*Sk] -> out[Ck, B*n]
__global__ __launch_bounds__(256)
void three_interp_kernel(const float* __restrict__ unknown, int ustride,
                         const float* __restrict__ known,
                         const float* __restrict__ kf, float* __restrict__ out,
                         int B, int n, int Sk, int Ck)
{
    int tid = blockIdx.x * blockDim.x + threadIdx.x;
    if (tid >= B * n) return;
    int b = tid / n;
    const float* u = unknown + (size_t)tid * ustride;
    float ux = u[0], uy = u[1], uz = u[2];
    const float* kp = known + (size_t)b * Sk * 3;
    float d0 = 3.4e38f, d1 = 3.4e38f, d2 = 3.4e38f;
    int i0 = 0, i1 = 0, i2 = 0;
    for (int j = 0; j < Sk; ++j) {
        float dx = kp[j * 3 + 0] - ux;
        float dy = kp[j * 3 + 1] - uy;
        float dz = kp[j * 3 + 2] - uz;
        float d = dx * dx + dy * dy + dz * dz;
        if (d < d0)      { d2 = d1; i2 = i1; d1 = d0; i1 = i0; d0 = d; i0 = j; }
        else if (d < d1) { d2 = d1; i2 = i1; d1 = d;  i1 = j; }
        else if (d < d2) { d2 = d;  i2 = j; }
    }
    float w0 = 1.0f / (d0 + 1e-8f);
    float w1 = 1.0f / (d1 + 1e-8f);
    float w2 = 1.0f / (d2 + 1e-8f);
    float ws = w0 + w1 + w2;
    w0 /= ws; w1 /= ws; w2 /= ws;
    size_t Pk = (size_t)B * Sk, Po = (size_t)B * n;
    for (int c = 0; c < Ck; ++c) {
        const float* row = kf + (size_t)c * Pk + (size_t)b * Sk;
        out[(size_t)c * Po + tid] = w0 * row[i0] + w1 * row[i1] + w2 * row[i2];
    }
}

// strided point-attribute transpose: dst[c*P + p] = src[p*stride + colOff + c]
__global__ __launch_bounds__(256)
void transpose_pts_kernel(const float* __restrict__ src, int stride, int colOff,
                          float* __restrict__ dst, int C, int P)
{
    int tid = blockIdx.x * blockDim.x + threadIdx.x;
    if (tid >= C * P) return;
    int c = tid / P;
    int p = tid % P;
    dst[tid] = src[(size_t)p * stride + colOff + c];
}

// row-block copy: dst[(rowOff+c)*P + p] = src[c*P + p]
__global__ __launch_bounds__(256)
void copy_rows_kernel(const float* __restrict__ src, float* __restrict__ dst,
                      int C, int P, int rowOff)
{
    int tid = blockIdx.x * blockDim.x + threadIdx.x;
    if (tid >= C * P) return;
    int c = tid / P;
    int p = tid % P;
    dst[(size_t)(rowOff + c) * P + p] = src[tid];
}

// channel-major [C,P] -> point-major [P,C]
__global__ __launch_bounds__(256)
void transpose_cp_kernel(const float* __restrict__ in, float* __restrict__ out,
                         int C, int P)
{
    int tid = blockIdx.x * blockDim.x + threadIdx.x;
    if (tid >= C * P) return;
    int p = tid / C;
    int c = tid % C;
    out[tid] = in[(size_t)c * P + p];
}

// log_softmax over the POINT axis (reference: axis=1 of [B,N,13]).
__global__ __launch_bounds__(256)
void log_softmax_kernel(const float* __restrict__ sem, float* __restrict__ pred,
                        float* __restrict__ logits, int B, int N)
{
    __shared__ float red[256];
    int b = blockIdx.x / 13;
    int c = blockIdx.x % 13;
    int t = threadIdx.x;
    const float* row = sem + (size_t)c * ((size_t)B * N) + (size_t)b * N;
    float mx = -3.4e38f;
    for (int j = t; j < N; j += 256) mx = fmaxf(mx, row[j]);
    red[t] = mx; __syncthreads();
    for (int s = 128; s > 0; s >>= 1) {
        if (t < s) red[t] = fmaxf(red[t], red[t + s]);
        __syncthreads();
    }
    mx = red[0];
    __syncthreads();
    float sum = 0.0f;
    for (int j = t; j < N; j += 256) sum += __expf(row[j] - mx);
    red[t] = sum; __syncthreads();
    for (int s = 128; s > 0; s >>= 1) {
        if (t < s) red[t] += red[t + s];
        __syncthreads();
    }
    float lse = mx + logf(red[0]);
    for (int j = t; j < N; j += 256) {
        float v = row[j];
        logits[((size_t)b * N + j) * 13 + c] = v;
        pred[((size_t)b * N + j) * 13 + c] = v - lse;
    }
}

// ===========================================================================
// Host orchestration
// ===========================================================================
struct Mlp { const float *W, *g, *b, *m, *v; };

static inline int g1(long long n) { return (int)((n + 255) / 256); }

template <int ACT, bool HAS_BIAS, bool HAS_BN>
static void run_conv(const float* W, const float* cb,
                     const float* g, const float* b, const float* m, const float* v,
                     const float* X, float* Y, int Cout, int Cin, int P,
                     _Float16* Ws, _Float16* Xs, hipStream_t stream)
{
    int KB = (Cin + 31) / 32;
    swizzle_w_kernel<<<g1((long long)Cout * KB * 32), 256, 0, stream>>>(W, Ws, Cout, Cin, KB);
    swizzle_x_kernel<<<g1((long long)KB * 8 * P), 256, 0, stream>>>(X, Xs, Cin, P, KB);
    conv_bn_act_wmma<ACT, HAS_BIAS, HAS_BN>
        <<<dim3(P / 64, (Cout + 15) / 16), 32, 0, stream>>>(
            Ws, cb, g, b, m, v, Xs, Y, Cout, KB, P);
}

#define CONV_RELU(LYR, X, Y, CO, CI, P)                                        \
    run_conv<1, false, true>((LYR).W, nullptr, (LYR).g, (LYR).b, (LYR).m,      \
                             (LYR).v, (X), (Y), (CO), (CI), (P), Wsb, Xsb, stream)

extern "C" void kernel_launch(void* const* d_in, const int* in_sizes, int n_in,
                              void* d_out, int out_size, void* d_ws, size_t ws_size,
                              hipStream_t stream)
{
    (void)in_sizes; (void)n_in; (void)out_size; (void)ws_size;
    const int B = 8, N0 = 4096;
    const int P0 = B * N0;                 // 32768

    // ---- inputs (jax pytree / alphabetical-dict-key flatten order) ----
    const float* Xpc = (const float*)d_in[0];
    auto L = [&](int base) -> Mlp {
        return { (const float*)d_in[base],     (const float*)d_in[base + 1],
                 (const float*)d_in[base + 2], (const float*)d_in[base + 3],
                 (const float*)d_in[base + 4] };
    };
    Mlp fp1_0 = L(1),  fp1_1 = L(6),  fp1_2 = L(11), fp1_3 = L(16);
    Mlp fp2_0 = L(21), fp2_1 = L(26);
    Mlp fp3_0 = L(31), fp3_1 = L(36);
    Mlp fp4_0 = L(41), fp4_1 = L(46);
    const float* h_b1 = (const float*)d_in[51];
    const float* h_b2 = (const float*)d_in[52];
    const float* h_b3 = (const float*)d_in[53];
    const float* bn1g = (const float*)d_in[54]; const float* bn1b = (const float*)d_in[55];
    const float* bn1m = (const float*)d_in[56]; const float* bn1v = (const float*)d_in[57];
    const float* bn2g = (const float*)d_in[58]; const float* bn2b = (const float*)d_in[59];
    const float* bn2m = (const float*)d_in[60]; const float* bn2v = (const float*)d_in[61];
    const float* h_w1 = (const float*)d_in[62];
    const float* h_w2 = (const float*)d_in[63];
    const float* h_w3 = (const float*)d_in[64];
    Mlp sa1_0 = L(65),  sa1_1 = L(70),  sa1_2 = L(75);
    Mlp sa2_0 = L(80),  sa2_1 = L(85),  sa2_2 = L(90);
    Mlp sa3_0 = L(95),  sa3_1 = L(100), sa3_2 = L(105);
    Mlp sa4_0 = L(110), sa4_1 = L(115), sa4_2 = L(120);

    // ---- workspace carve (~200 MB); all offsets multiples of 64 floats ----
    float* w = (float*)d_ws;
    size_t off = 0;
    auto alloc = [&](size_t n) { float* p = w + off; off += n; return p; };
    float* bufA      = alloc(16777216);        // GEMM ping
    float* bufB      = alloc(16777216);        // GEMM pong
    _Float16* Xsb    = (_Float16*)alloc(8388608);   // 16M halves: swizzled X
    _Float16* Wsb    = (_Float16*)alloc(131072);    // 256K halves: swizzled W
    float* l0_points = alloc((size_t)6 * P0);
    float* l0_in     = alloc((size_t)9 * P0);
    float* l1_xyz    = alloc((size_t)B * 1024 * 3);
    float* l1_points = alloc((size_t)64 * B * 1024);
    float* l2_xyz    = alloc((size_t)B * 256 * 3);
    float* l2_points = alloc((size_t)128 * B * 256);
    float* l3_xyz    = alloc((size_t)B * 64 * 3);
    float* l3_points = alloc((size_t)256 * B * 64);
    float* l4_points = alloc((size_t)512 * B);
    float* fp4_out   = alloc((size_t)256 * B * 64);
    float* fp3_out   = alloc((size_t)256 * B * 256);
    float* fp2_out   = alloc((size_t)128 * B * 1024);
    float* fp1_out   = alloc((size_t)128 * P0);
    float* sem3      = alloc((size_t)13 * P0);
    int* fps_idx  = (int*)(w + off); off += (size_t)B * 1024;
    int* ball_idx = (int*)(w + off); off += (size_t)B * 1024 * 32;

    // ---- outputs ----
    float* out        = (float*)d_out;
    float* out_pf     = out;                               // [B,4096,128]
    float* out_gf     = out + (size_t)P0 * 128;            // [B,512]
    float* out_pred   = out_gf + (size_t)B * 512;          // [B,4096,13]
    float* out_logits = out_pred + (size_t)P0 * 13;        // [B,4096,13]

    // ---- l0 features ----
    transpose_pts_kernel<<<g1((long long)6 * P0), 256, 0, stream>>>(Xpc, 9, 3, l0_points, 6, P0);
    transpose_pts_kernel<<<g1((long long)3 * P0), 256, 0, stream>>>(Xpc, 9, 0, l0_in, 3, P0);
    copy_rows_kernel<<<g1((long long)6 * P0), 256, 0, stream>>>(l0_points, l0_in, 6, P0, 3);

    // ================= SA1: N=4096 -> S=1024, K=32, r=0.1, ch 9->32->32->64
    {
        const int N = 4096, S = 1024, K = 32, Cf = 6;
        const int P = B * S * K;  // 262144
        fps_kernel<<<B, 256, 0, stream>>>(Xpc, 9, N, S, fps_idx);
        gather_xyz_kernel<<<g1(B * S), 256, 0, stream>>>(Xpc, 9, fps_idx, l1_xyz, B, N, S);
        ball_query_kernel<<<g1(B * S), 256, 0, stream>>>(Xpc, 9, l1_xyz, 0.1f * 0.1f, K, ball_idx, B, N, S);
        group_kernel<<<g1(P), 256, 0, stream>>>(Xpc, 9, l1_xyz, l0_points, ball_idx, bufA, B, N, S, K, Cf);
        CONV_RELU(sa1_0, bufA, bufB, 32, 9, P);
        CONV_RELU(sa1_1, bufB, bufA, 32, 32, P);
        CONV_RELU(sa1_2, bufA, bufB, 64, 32, P);
        maxpool_kernel<<<g1((long long)64 * B * S), 256, 0, stream>>>(bufB, l1_points, 64, B * S, K);
    }
    // ================= SA2: N=1024 -> S=256, K=64, r=0.2, ch 67->64->64->128
    {
        const int N = 1024, S = 256, K = 64, Cf = 64;
        const int P = B * S * K;  // 131072
        fps_kernel<<<B, 256, 0, stream>>>(l1_xyz, 3, N, S, fps_idx);
        gather_xyz_kernel<<<g1(B * S), 256, 0, stream>>>(l1_xyz, 3, fps_idx, l2_xyz, B, N, S);
        ball_query_kernel<<<g1(B * S), 256, 0, stream>>>(l1_xyz, 3, l2_xyz, 0.2f * 0.2f, K, ball_idx, B, N, S);
        group_kernel<<<g1(P), 256, 0, stream>>>(l1_xyz, 3, l2_xyz, l1_points, ball_idx, bufA, B, N, S, K, Cf);
        CONV_RELU(sa2_0, bufA, bufB, 64, 67, P);
        CONV_RELU(sa2_1, bufB, bufA, 64, 64, P);
        CONV_RELU(sa2_2, bufA, bufB, 128, 64, P);
        maxpool_kernel<<<g1((long long)128 * B * S), 256, 0, stream>>>(bufB, l2_points, 128, B * S, K);
    }
    // ================= SA3: N=256 -> S=64, K=128, r=0.4, ch 131->128->128->256
    {
        const int N = 256, S = 64, K = 128, Cf = 128;
        const int P = B * S * K;  // 65536
        fps_kernel<<<B, 256, 0, stream>>>(l2_xyz, 3, N, S, fps_idx);
        gather_xyz_kernel<<<g1(B * S), 256, 0, stream>>>(l2_xyz, 3, fps_idx, l3_xyz, B, N, S);
        ball_query_kernel<<<g1(B * S), 256, 0, stream>>>(l2_xyz, 3, l3_xyz, 0.4f * 0.4f, K, ball_idx, B, N, S);
        group_kernel<<<g1(P), 256, 0, stream>>>(l2_xyz, 3, l3_xyz, l2_points, ball_idx, bufA, B, N, S, K, Cf);
        CONV_RELU(sa3_0, bufA, bufB, 128, 131, P);
        CONV_RELU(sa3_1, bufB, bufA, 128, 128, P);
        CONV_RELU(sa3_2, bufA, bufB, 256, 128, P);
        maxpool_kernel<<<g1((long long)256 * B * S), 256, 0, stream>>>(bufB, l3_points, 256, B * S, K);
    }
    // ================= SA4 (group-all): 64 pts/batch, ch 259->256->256->512
    {
        const int P = B * 64;  // 512
        transpose_pts_kernel<<<g1((long long)3 * P), 256, 0, stream>>>(l3_xyz, 3, 0, bufA, 3, P);
        copy_rows_kernel<<<g1((long long)256 * P), 256, 0, stream>>>(l3_points, bufA, 256, P, 3);
        CONV_RELU(sa4_0, bufA, bufB, 256, 259, P);
        CONV_RELU(sa4_1, bufB, bufA, 256, 256, P);
        CONV_RELU(sa4_2, bufA, bufB, 512, 256, P);
        maxpool_kernel<<<g1((long long)512 * B), 256, 0, stream>>>(bufB, l4_points, 512, B, 64);
    }
    // ================= FP4: broadcast l4 -> l3 points, ch 768->256->256
    {
        const int P = B * 64;
        broadcast_kernel<<<g1((long long)512 * P), 256, 0, stream>>>(l4_points, bufA, 512, B, 64);
        copy_rows_kernel<<<g1((long long)256 * P), 256, 0, stream>>>(l3_points, bufA, 256, P, 512);
        CONV_RELU(fp4_0, bufA, bufB, 256, 768, P);
        CONV_RELU(fp4_1, bufB, fp4_out, 256, 256, P);
    }
    // ================= FP3: l3(64) -> l2(256), ch 384->256->256
    {
        const int n = 256, Sk = 64, P = B * n;
        three_interp_kernel<<<g1(B * n), 256, 0, stream>>>(l2_xyz, 3, l3_xyz, fp4_out, bufA, B, n, Sk, 256);
        copy_rows_kernel<<<g1((long long)128 * P), 256, 0, stream>>>(l2_points, bufA, 128, P, 256);
        CONV_RELU(fp3_0, bufA, bufB, 256, 384, P);
        CONV_RELU(fp3_1, bufB, fp3_out, 256, 256, P);
    }
    // ================= FP2: l2(256) -> l1(1024), ch 320->256->128
    {
        const int n = 1024, Sk = 256, P = B * n;
        three_interp_kernel<<<g1(B * n), 256, 0, stream>>>(l1_xyz, 3, l2_xyz, fp3_out, bufA, B, n, Sk, 256);
        copy_rows_kernel<<<g1((long long)64 * P), 256, 0, stream>>>(l1_points, bufA, 64, P, 256);
        CONV_RELU(fp2_0, bufA, bufB, 256, 320, P);
        CONV_RELU(fp2_1, bufB, fp2_out, 128, 256, P);
    }
    // ================= FP1: l1(1024) -> l0(4096), ch 137->128 x4
    {
        const int n = 4096, Sk = 1024, P = B * n;  // 32768
        three_interp_kernel<<<g1(B * n), 256, 0, stream>>>(Xpc, 9, l1_xyz, fp2_out, bufA, B, n, Sk, 128);
        copy_rows_kernel<<<g1((long long)9 * P), 256, 0, stream>>>(l0_in, bufA, 9, P, 128);
        CONV_RELU(fp1_0, bufA, bufB, 128, 137, P);
        CONV_RELU(fp1_1, bufB, bufA, 128, 128, P);
        CONV_RELU(fp1_2, bufA, bufB, 128, 128, P);
        CONV_RELU(fp1_3, bufB, fp1_out, 128, 128, P);
    }
    // ================= Head: 128->128 (bias+bn+leaky), 128->64, 64->13 (bias)
    {
        const int P = P0;
        run_conv<2, true, true>(h_w1, h_b1, bn1g, bn1b, bn1m, bn1v,
                                fp1_out, bufA, 128, 128, P, Wsb, Xsb, stream);
        run_conv<2, true, true>(h_w2, h_b2, bn2g, bn2b, bn2m, bn2v,
                                bufA, bufB, 64, 128, P, Wsb, Xsb, stream);
        run_conv<0, true, false>(h_w3, h_b3, nullptr, nullptr, nullptr, nullptr,
                                 bufB, sem3, 13, 64, P, Wsb, Xsb, stream);
    }
    // ================= Outputs
    transpose_cp_kernel<<<g1((long long)128 * P0), 256, 0, stream>>>(fp1_out, out_pf, 128, P0);
    transpose_cp_kernel<<<g1((long long)512 * B), 256, 0, stream>>>(l4_points, out_gf, 512, B);
    log_softmax_kernel<<<B * 13, 256, 0, stream>>>(sem3, out_pred, out_logits, B, N0);
}
